// ChannelAttention_71124658422014
// MI455X (gfx1250) — compile-verified, matrix operand order checked by reading, not run on hardware
//
#include <hip/hip_runtime.h>

#define B_ 32
#define S_ 4096
#define E_ 64
#define SE_ (S_ * E_)

typedef __attribute__((ext_vector_type(8)))  int          v8i;
typedef __attribute__((ext_vector_type(8)))  float        v8f;
typedef __attribute__((ext_vector_type(16))) __bf16       v16bf;
typedef __attribute__((ext_vector_type(4)))  unsigned int u32x4;
typedef __attribute__((ext_vector_type(2)))  unsigned int u32x2;
typedef __attribute__((ext_vector_type(4)))  float        f32x4;

__device__ __forceinline__ unsigned short f2bf(float f) {
    unsigned int u = __float_as_uint(f);
    return (unsigned short)((u + 0x7fffu + ((u >> 16) & 1u)) >> 16);
}
__device__ __forceinline__ float gelu_exact(float x) {
    return 0.5f * x * (1.0f + erff(x * 0.70710678118654752f));
}

__device__ __forceinline__ v16bf ld_frag_bf16(const unsigned short* p, int lane, int ld) {
    int m  = lane & 15;
    int ks = (lane >> 4) << 3;
    const unsigned short* r = p + (size_t)m * ld + ks;
    union { u32x4 d[2]; v16bf v; } f;
    f.d[0] = *(const u32x4*)(r);
    f.d[1] = *(const u32x4*)(r + 16);
    return f.v;
}

__device__ __forceinline__ v8i ld_frag_i8(const void* pv, int lane, int ld) {
    const char* p = (const char*)pv;
    int m  = lane & 15;
    int ks = (lane >> 4) << 3;
    const char* r = p + m * ld + ks;
    union { u32x2 d[4]; v8i v; } f;
    f.d[0] = *(const u32x2*)(r + 0);
    f.d[1] = *(const u32x2*)(r + 16);
    f.d[2] = *(const u32x2*)(r + 32);
    f.d[3] = *(const u32x2*)(r + 48);
    return f.v;
}

__device__ __forceinline__ void layernorm_row(float* row, const float* g, const float* bb) {
    float m = 0.f;
    for (int e = 0; e < 64; ++e) m += row[e];
    m *= (1.f / 64.f);
    float v = 0.f;
    for (int e = 0; e < 64; ++e) { float d = row[e] - m; v += d * d; }
    v *= (1.f / 64.f);
    float rs = rsqrtf(v + 1e-5f);
    for (int e = 0; e < 64; ++e) row[e] = (row[e] - m) * rs * g[e] + bb[e];
}

__device__ __forceinline__ float quant_row(const float* row, unsigned char* qrow) {
    float ma = 0.f;
    for (int e = 0; e < 64; ++e) ma = fmaxf(ma, fabsf(row[e]));
    ma = fmaxf(ma, 1e-8f);
    float gamma = 127.f / ma;
    for (int e = 0; e < 64; ++e) {
        float q = rintf(row[e] * gamma);
        q = fminf(fmaxf(q, -128.f), 127.f);
        qrow[e] = (unsigned char)(signed char)(int)q;
    }
    return ma * (1.f / 127.f);
}

__global__ __launch_bounds__(256) void prep_w_kernel(
    const float* __restrict__ qw, const float* __restrict__ kw, const float* __restrict__ vw,
    const float* __restrict__ f1w, const float* __restrict__ f2w,
    signed char* __restrict__ wq5, float* __restrict__ betas)
{
    int blk = blockIdx.x, tid = threadIdx.x;
    const float* w = (blk == 0) ? qw : (blk == 1) ? kw : (blk == 2) ? vw : (blk == 3) ? f1w : f2w;
    __shared__ float red[256];
    float s = 0.f;
    for (int i = tid; i < 4096; i += 256) s += fabsf(w[i]);
    red[tid] = s;
    __syncthreads();
    for (int st = 128; st > 0; st >>= 1) {
        if (tid < st) red[tid] += red[tid + st];
        __syncthreads();
    }
    float beta = fmaxf(red[0] * (1.f / 4096.f), 1e-8f);
    if (tid == 0) betas[blk] = beta;
    float inv = 1.f / beta;
    signed char* dst = wq5 + blk * 4096;
    for (int i = tid; i < 4096; i += 256) {
        float q = rintf(w[i] * inv);
        q = fminf(fmaxf(q, -1.f), 1.f);
        dst[i] = (signed char)(int)q;
    }
}

__device__ __forceinline__ void conv_pass(const float (*in)[66], float (*out)[66],
                                          const float* w, float bias, int tid) {
    float w0 = w[0], w1 = w[1], w2 = w[2], w3 = w[3], w4 = w[4],
          w5 = w[5], w6 = w[6], w7 = w[7], w8 = w[8];
    for (int i = tid; i < 4096; i += 256) {
        int r = i >> 6, c = i & 63;
        float s = w0 * in[r][c]     + w1 * in[r][c + 1]     + w2 * in[r][c + 2]
                + w3 * in[r + 1][c] + w4 * in[r + 1][c + 1] + w5 * in[r + 1][c + 2]
                + w6 * in[r + 2][c] + w7 * in[r + 2][c + 1] + w8 * in[r + 2][c + 2];
        out[r + 1][c + 1] = gelu_exact(s + bias);
    }
}

__global__ __launch_bounds__(256) void conv_kernel(
    const float* __restrict__ it2,
    const float* __restrict__ c1w, const float* __restrict__ c1b,
    const float* __restrict__ c2w, const float* __restrict__ c2b,
    const float* __restrict__ c3w, const float* __restrict__ c3b,
    float* __restrict__ it_g)
{
    __shared__ float A[66][66];
    __shared__ float Bf[66][66];
    int b = blockIdx.x, tid = threadIdx.x;
    for (int i = tid; i < 66 * 66; i += 256) { A[i / 66][i % 66] = 0.f; Bf[i / 66][i % 66] = 0.f; }
    __syncthreads();
    const float* src = it2 + (size_t)b * 1024;
    for (int i = tid; i < 4096; i += 256) {
        int r = i >> 6, c = i & 63;
        float sr = r * 0.5f - 0.25f, sc = c * 0.5f - 0.25f;
        int r0 = (int)floorf(sr), c0 = (int)floorf(sc);
        float fr = sr - (float)r0, fc = sc - (float)c0;
        int r1 = min(max(r0 + 1, 0), 31), c1 = min(max(c0 + 1, 0), 31);
        r0 = min(max(r0, 0), 31); c0 = min(max(c0, 0), 31);
        float v = (1.f - fr) * ((1.f - fc) * src[r0 * 32 + c0] + fc * src[r0 * 32 + c1])
                + fr * ((1.f - fc) * src[r1 * 32 + c0] + fc * src[r1 * 32 + c1]);
        A[r + 1][c + 1] = v;
    }
    __syncthreads();
    conv_pass(A, Bf, c1w, c1b[0], tid); __syncthreads();
    conv_pass(Bf, A, c2w, c2b[0], tid); __syncthreads();
    conv_pass(A, Bf, c3w, c3b[0], tid); __syncthreads();
    float* dst = it_g + (size_t)b * 4096;
    for (int i = tid; i < 4096; i += 256) dst[i] = Bf[(i >> 6) + 1][(i & 63) + 1];
}

__global__ __launch_bounds__(256) void qkv_kernel(
    const float* __restrict__ x, const float* __restrict__ ln1g, const float* __restrict__ ln1b,
    const signed char* __restrict__ wq5, const float* __restrict__ betas,
    const float* __restrict__ qb_, const float* __restrict__ kb_, const float* __restrict__ vb_,
    float* __restrict__ xn_g,
    unsigned short* __restrict__ q_g, unsigned short* __restrict__ k_g, unsigned short* __restrict__ v_g)
{
    __shared__ __align__(16) float sx[128 * 64];
    __shared__ __align__(16) unsigned char sq[128 * 64];
    __shared__ float sscale[128];
    int b  = blockIdx.x >> 5;
    int s0 = (blockIdx.x & 31) * 128;
    int tid = threadIdx.x;
    size_t base = (size_t)b * SE_ + (size_t)s0 * 64;

    const f32x4* xin = (const f32x4*)(x + base);
    f32x4* sx4 = (f32x4*)sx;
    for (int i = tid; i < 128 * 16; i += 256) sx4[i] = xin[i];
    __syncthreads();

    if (tid < 128) {
        float* row = sx + tid * 64;
        layernorm_row(row, ln1g, ln1b);
        sscale[tid] = quant_row(row, sq + tid * 64);
    }
    __syncthreads();

    f32x4* xng = (f32x4*)(xn_g + base);
    for (int i = tid; i < 128 * 16; i += 256) xng[i] = sx4[i];

    int wave = tid >> 5, lane = tid & 31;
    int m0 = wave * 16;
    v8i a = ld_frag_i8(sq + m0 * 64, lane, 64);
    int n = lane & 15;
    int mhi = (lane >> 4) << 3;

    for (int L = 0; L < 3; ++L) {
        const signed char* wl = wq5 + L * 4096;
        float beta = betas[L];
        const float* bias = (L == 0) ? qb_ : (L == 1) ? kb_ : vb_;
        unsigned short* og = ((L == 0) ? q_g : (L == 1) ? k_g : v_g) + (size_t)b * SE_;
        float rs[8];
        #pragma unroll
        for (int vv = 0; vv < 8; ++vv) rs[vv] = beta * sscale[m0 + mhi + vv];
        for (int nt = 0; nt < 4; ++nt) {
            int n0 = nt * 16;
            v8i bf = ld_frag_i8(wl + n0 * 64, lane, 64);
            v8i cz = {0, 0, 0, 0, 0, 0, 0, 0};
            v8i d = __builtin_amdgcn_wmma_i32_16x16x64_iu8(true, a, true, bf, cz, false, false);
            float bv = bias[n0 + n];
            #pragma unroll
            for (int vv = 0; vv < 8; ++vv) {
                int rr = m0 + mhi + vv;
                float y = (float)d[vv] * rs[vv] + bv;
                og[(size_t)(s0 + rr) * 64 + n0 + n] = f2bf(y);
            }
        }
    }
}

__global__ __launch_bounds__(512) void scores_kernel(
    const unsigned short* __restrict__ q_g, const unsigned short* __restrict__ k_g,
    const float* __restrict__ it_g, unsigned short* __restrict__ attn_g)
{
    __shared__ __align__(16) unsigned short sQ[64][136];
    __shared__ __align__(16) unsigned short sK[64][136];
    __shared__ float ssc[64][65];
    int b = blockIdx.x;
    int tid = threadIdx.x, wave = tid >> 5, lane = tid & 31;
    const unsigned short* Q = q_g + (size_t)b * SE_;
    const unsigned short* K = k_g + (size_t)b * SE_;
    int tm = wave >> 2, tn = wave & 3;
    v8f acc = {0.f, 0.f, 0.f, 0.f, 0.f, 0.f, 0.f, 0.f};

    int lrow = tid >> 3;
    int lseg = (tid & 7) * 16;
    const unsigned short* qrp = Q + (size_t)lrow * 4096 + lseg;
    const unsigned short* krp = K + (size_t)lrow * 4096 + lseg;

    for (int kb = 0; kb < 4096; kb += 128) {
        *(u32x4*)&sQ[lrow][lseg]     = *(const u32x4*)(qrp);
        *(u32x4*)&sQ[lrow][lseg + 8] = *(const u32x4*)(qrp + 8);
        *(u32x4*)&sK[lrow][lseg]     = *(const u32x4*)(krp);
        *(u32x4*)&sK[lrow][lseg + 8] = *(const u32x4*)(krp + 8);
        if (kb + 128 < 4096) {
            __builtin_prefetch(qrp + 128, 0, 1);
            __builtin_prefetch(krp + 128, 0, 1);
        }
        qrp += 128;
        krp += 128;
        __syncthreads();
        #pragma unroll
        for (int kk = 0; kk < 128; kk += 32) {
            v16bf a  = ld_frag_bf16(&sQ[tm * 16][kk], lane, 136);
            v16bf bb = ld_frag_bf16(&sK[tn * 16][kk], lane, 136);
            acc = __builtin_amdgcn_wmma_f32_16x16x32_bf16(false, a, false, bb, (short)0, acc, false, false);
        }
        __syncthreads();
    }

    int n = lane & 15, mhi = (lane >> 4) << 3;
    const float* itb = it_g + (size_t)b * 4096;
    #pragma unroll
    for (int vv = 0; vv < 8; ++vv) {
        int r = tm * 16 + mhi + vv, c = tn * 16 + n;
        ssc[r][c] = acc[vv] * 0.125f + itb[r * 64 + c];
    }
    __syncthreads();
    if (tid < 64) {
        float mx = -3.4e38f;
        for (int c = 0; c < 64; ++c) mx = fmaxf(mx, ssc[tid][c]);
        float sum = 0.f;
        for (int c = 0; c < 64; ++c) { float e = expf(ssc[tid][c] - mx); ssc[tid][c] = e; sum += e; }
        float inv = 1.f / sum;
        unsigned short* arow = attn_g + (size_t)b * 4096 + tid * 64;
        for (int c = 0; c < 64; ++c) arow[c] = f2bf(ssc[tid][c] * inv);
    }
}

__global__ __launch_bounds__(256) void av_kernel(
    const unsigned short* __restrict__ v_g, const unsigned short* __restrict__ attn_g,
    const float* __restrict__ xn_g, float* __restrict__ ao_g)
{
    __shared__ __align__(16) unsigned short sVT[64][72];
    __shared__ __align__(16) unsigned short sAT[64][72];
    int b  = blockIdx.x >> 6;
    int s0 = (blockIdx.x & 63) * 64;
    int tid = threadIdx.x, wave = tid >> 5, lane = tid & 31;
    const unsigned short* V  = v_g + (size_t)b * SE_;
    const unsigned short* At = attn_g + (size_t)b * 4096;

    for (int i = tid; i < 4096; i += 256) {
        int f = i >> 6, c = i & 63;
        sVT[c][f] = V[(size_t)f * 4096 + s0 + c];
        sAT[f][c] = At[f * 64 + c];
    }
    __syncthreads();

    int n = lane & 15, mhi = (lane >> 4) << 3;
    for (int t = wave * 2; t < wave * 2 + 2; ++t) {
        int mt = t >> 2, nt = t & 3;
        int n0 = nt * 16;
        v8f acc = {0.f, 0.f, 0.f, 0.f, 0.f, 0.f, 0.f, 0.f};
        #pragma unroll
        for (int k0 = 0; k0 < 64; k0 += 32) {
            v16bf a  = ld_frag_bf16(&sVT[mt * 16][k0], lane, 72);
            v16bf bb = ld_frag_bf16(&sAT[n0][k0], lane, 72);
            acc = __builtin_amdgcn_wmma_f32_16x16x32_bf16(false, a, false, bb, (short)0, acc, false, false);
        }
        #pragma unroll
        for (int vv = 0; vv < 8; ++vv) {
            size_t idx = (size_t)b * SE_ + (size_t)(s0 + mt * 16 + mhi + vv) * 64 + n0 + n;
            ao_g[idx] = acc[vv] - xn_g[idx];
        }
    }
}

__global__ __launch_bounds__(256) void epilogue_kernel(
    const float* __restrict__ ao_g, const float* __restrict__ x_g,
    const float* __restrict__ ln2g, const float* __restrict__ ln2b,
    const float* __restrict__ ln3g, const float* __restrict__ ln3b,
    const float* __restrict__ ln4g, const float* __restrict__ ln4b,
    const signed char* __restrict__ wq5, const float* __restrict__ betas,
    const float* __restrict__ f1b_, const float* __restrict__ f2b_,
    float* __restrict__ out_g)
{
    __shared__ __align__(16) float sh[128 * 64];
    __shared__ __align__(16) float sr2[128 * 64];
    __shared__ __align__(16) unsigned char sq[128 * 64];
    __shared__ float sscale[128];
    int b  = blockIdx.x >> 5;
    int s0 = (blockIdx.x & 31) * 128;
    int tid = threadIdx.x;
    size_t base = (size_t)b * SE_ + (size_t)s0 * 64;

    const f32x4* aoin = (const f32x4*)(ao_g + base);
    const f32x4* xin  = (const f32x4*)(x_g + base);
    f32x4* sh4  = (f32x4*)sh;
    f32x4* sr24 = (f32x4*)sr2;
    for (int i = tid; i < 128 * 16; i += 256) { sh4[i] = aoin[i]; sr24[i] = xin[i]; }
    __syncthreads();

    if (tid < 128) {
        float* hrow = sh + tid * 64;
        layernorm_row(hrow, ln2g, ln2b);
        float* r2 = sr2 + tid * 64;
        for (int e = 0; e < 64; ++e) { float t = r2[e] + hrow[e]; r2[e] = t; hrow[e] = t; }
        layernorm_row(hrow, ln3g, ln3b);
        sscale[tid] = quant_row(hrow, sq + tid * 64);
    }
    __syncthreads();

    int wave = tid >> 5, lane = tid & 31;
    int m0 = wave * 16;
    int n = lane & 15, mhi = (lane >> 4) << 3;

    {
        v8i a = ld_frag_i8(sq + m0 * 64, lane, 64);
        const signed char* w1 = wq5 + 3 * 4096;
        float beta1 = betas[3];
        float rs[8];
        #pragma unroll
        for (int vv = 0; vv < 8; ++vv) rs[vv] = beta1 * sscale[m0 + mhi + vv];
        for (int nt = 0; nt < 4; ++nt) {
            int n0 = nt * 16;
            v8i bf = ld_frag_i8(w1 + n0 * 64, lane, 64);
            v8i cz = {0, 0, 0, 0, 0, 0, 0, 0};
            v8i d = __builtin_amdgcn_wmma_i32_16x16x64_iu8(true, a, true, bf, cz, false, false);
            float bv = f1b_[n0 + n];
            #pragma unroll
            for (int vv = 0; vv < 8; ++vv) {
                int rr = m0 + mhi + vv;
                sh[rr * 64 + n0 + n] = gelu_exact((float)d[vv] * rs[vv] + bv);
            }
        }
    }
    __syncthreads();

    if (tid < 128) {
        float* row = sh + tid * 64;
        layernorm_row(row, ln4g, ln4b);
        sscale[tid] = quant_row(row, sq + tid * 64);
    }
    __syncthreads();

    {
        v8i a = ld_frag_i8(sq + m0 * 64, lane, 64);
        const signed char* w2 = wq5 + 4 * 4096;
        float beta2 = betas[4];
        float rs[8];
        #pragma unroll
        for (int vv = 0; vv < 8; ++vv) rs[vv] = beta2 * sscale[m0 + mhi + vv];
        for (int nt = 0; nt < 4; ++nt) {
            int n0 = nt * 16;
            v8i bf = ld_frag_i8(w2 + n0 * 64, lane, 64);
            v8i cz = {0, 0, 0, 0, 0, 0, 0, 0};
            v8i d = __builtin_amdgcn_wmma_i32_16x16x64_iu8(true, a, true, bf, cz, false, false);
            float bv = f2b_[n0 + n];
            #pragma unroll
            for (int vv = 0; vv < 8; ++vv) {
                int rr = m0 + mhi + vv;
                float y = (float)d[vv] * rs[vv] + bv + sr2[rr * 64 + n0 + n];
                out_g[base + (size_t)rr * 64 + n0 + n] = y;
            }
        }
    }
}

extern "C" void kernel_launch(void* const* d_in, const int* in_sizes, int n_in,
                              void* d_out, int out_size, void* d_ws, size_t ws_size,
                              hipStream_t stream) {
    (void)in_sizes; (void)n_in; (void)out_size; (void)ws_size;
    const float* x    = (const float*)d_in[0];
    const float* it2  = (const float*)d_in[1];
    const float* qw   = (const float*)d_in[2];
    const float* qb   = (const float*)d_in[3];
    const float* kw   = (const float*)d_in[4];
    const float* kb   = (const float*)d_in[5];
    const float* vw   = (const float*)d_in[6];
    const float* vb   = (const float*)d_in[7];
    const float* f1w  = (const float*)d_in[8];
    const float* f1b  = (const float*)d_in[9];
    const float* f2w  = (const float*)d_in[10];
    const float* f2b  = (const float*)d_in[11];
    const float* c1w  = (const float*)d_in[12];
    const float* c1b  = (const float*)d_in[13];
    const float* c2w  = (const float*)d_in[14];
    const float* c2b  = (const float*)d_in[15];
    const float* c3w  = (const float*)d_in[16];
    const float* c3b  = (const float*)d_in[17];
    const float* ln1g = (const float*)d_in[18];
    const float* ln1b = (const float*)d_in[19];
    const float* ln2g = (const float*)d_in[20];
    const float* ln2b = (const float*)d_in[21];
    const float* ln3g = (const float*)d_in[22];
    const float* ln3b = (const float*)d_in[23];
    const float* ln4g = (const float*)d_in[24];
    const float* ln4b = (const float*)d_in[25];
    float* out = (float*)d_out;

    char* w = (char*)d_ws;
    size_t o = 0;
    auto carve = [&](size_t bytes) -> char* {
        char* p = w + o;
        o += bytes;
        o = (o + 255) & ~(size_t)255;
        return p;
    };
    signed char*    wq5    = (signed char*)   carve(5 * 4096);
    float*          betas  = (float*)         carve(5 * sizeof(float));
    float*          it_g   = (float*)         carve((size_t)B_ * 4096 * 4);
    unsigned short* attn_g = (unsigned short*)carve((size_t)B_ * 4096 * 2);
    float*          xn_g   = (float*)         carve((size_t)B_ * SE_ * 4);
    unsigned short* q_g    = (unsigned short*)carve((size_t)B_ * SE_ * 2);
    unsigned short* k_g    = (unsigned short*)carve((size_t)B_ * SE_ * 2);
    unsigned short* v_g    = (unsigned short*)carve((size_t)B_ * SE_ * 2);
    float*          ao_g   = (float*)         carve((size_t)B_ * SE_ * 4);

    prep_w_kernel<<<5, 256, 0, stream>>>(qw, kw, vw, f1w, f2w, wq5, betas);
    conv_kernel<<<B_, 256, 0, stream>>>(it2, c1w, c1b, c2w, c2b, c3w, c3b, it_g);
    qkv_kernel<<<B_ * (S_ / 128), 256, 0, stream>>>(x, ln1g, ln1b, wq5, betas,
                                                    qb, kb, vb, xn_g, q_g, k_g, v_g);
    scores_kernel<<<B_, 512, 0, stream>>>(q_g, k_g, it_g, attn_g);
    av_kernel<<<B_ * (S_ / 64), 256, 0, stream>>>(v_g, attn_g, xn_g, ao_g);
    epilogue_kernel<<<B_ * (S_ / 128), 256, 0, stream>>>(ao_g, x,
                                                         ln2g, ln2b, ln3g, ln3b, ln4g, ln4b,
                                                         wq5, betas, f1b, f2b, out);
}
